// GCN_5875515261182
// MI455X (gfx1250) — matrix operand order repaired and checked by reading, not compiled, not run on hardware
//
#include <hip/hip_runtime.h>

// ---------------------------------------------------------------------------
// GCN forward for MI455X (gfx1250, wave32).
// deg -> dinv -> edge norms -> 3x [TDM-staged WMMA bf16 GEMM, edge scatter
// with f32 HW atomics, bias+act(+bf16 copy)] -> mean pool -> linear head.
// GEMM: block = 256 thr (8 waves), 64x128 output stripe.
//   - Operands pre-converted to bf16 in global (W transposed [n][k]).
//   - Tiles DMA'd into LDS by TENSOR_LOAD_TO_LDS (TDM), s_wait_tensorcnt.
//   - B frags loaded once per wave into registers, reused across 4 row tiles.
//   - 16 x v_wmma_f32_16x16x32_bf16 per wave; block-uniform store guards.
// ---------------------------------------------------------------------------

typedef __attribute__((ext_vector_type(16))) __bf16 v16bf;
typedef __attribute__((ext_vector_type(8)))  __bf16 v8bf;
typedef __attribute__((ext_vector_type(8)))  float  v8f;
typedef __attribute__((ext_vector_type(4)))  unsigned int v4u;
typedef __attribute__((ext_vector_type(8)))  int v8i;
typedef __attribute__((ext_vector_type(4)))  int v4i;

__device__ __forceinline__ float fatomic_add(float* p, float v) {
  return unsafeAtomicAdd(p, v);   // global_atomic_add_f32, no CAS loop
}

__device__ __forceinline__ v16bf cat8(v8bf lo, v8bf hi) {
  return __builtin_shufflevector(lo, hi, 0, 1, 2, 3, 4, 5, 6, 7,
                                         8, 9, 10, 11, 12, 13, 14, 15);
}

static __host__ __device__ __forceinline__ int cdiv(int a, int b) { return (a + b - 1) / b; }

// ---- TDM: 2D bf16 tile (tile_d0=128 elems/row) global -> LDS ---------------
// Descriptor per CDNA5 ISA section 8.3/8.4. tensor_dim0 = 128 elements,
// row stride = 128 elements; rows beyond tensor_d1 read as zero.
__device__ __forceinline__ void tdm_load_2d_bf16(unsigned lds_off,
                                                 const void* gaddr,
                                                 unsigned tensor_d1,
                                                 unsigned tile_d1) {
  unsigned long long ga = (unsigned long long)gaddr;
  v4u g0;
  g0[0] = 1u;                                              // count=1, user D#
  g0[1] = lds_off;                                         // LDS byte address
  g0[2] = (unsigned)ga;                                    // global_addr[31:0]
  g0[3] = (unsigned)((ga >> 32) & 0x01FFFFFFu) | (2u << 30); // addr[56:32] | type=2
  v8i g1;
  g1[0] = (int)(1u << 16);                                 // data_size=1 -> 2 bytes
  g1[1] = (int)(128u << 16);                               // tensor_dim0 = 128
  g1[2] = (int)((tensor_d1 & 0xFFFFu) << 16);              // tensor_dim1 lo16
  g1[3] = (int)((tensor_d1 >> 16) & 0xFFFFu) | (int)(128u << 16); // dim1 hi | tile_dim0=128
  g1[4] = (int)(tile_d1 & 0xFFFFu);                        // tile_dim1 (tile_dim2=0)
  g1[5] = 128;                                             // tensor_dim0_stride lo32
  g1[6] = 0;
  g1[7] = 0;
  v4i z4 = {0, 0, 0, 0};
  v8i z8 = {0, 0, 0, 0, 0, 0, 0, 0};
  __builtin_amdgcn_tensor_load_to_lds(g0, g1, z4, z4, z8, 0);
}

// ---- degree: deg[dst] += ew over real edges --------------------------------
__global__ void k_deg(const int* __restrict__ dst, const float* __restrict__ ew,
                      float* __restrict__ deg, int E) {
  int e = blockIdx.x * blockDim.x + threadIdx.x;
  if (e < E) fatomic_add(&deg[dst[e]], ew[e]);
}

// ---- dinv for both self-loop fill values (2.0 layer1, 1.0 otherwise) -------
__global__ void k_dinv(const float* __restrict__ deg, float* __restrict__ dinv1,
                       float* __restrict__ dinv, int N) {
  int i = blockIdx.x * blockDim.x + threadIdx.x;
  if (i >= N) return;
  float d  = deg[i];
  float d1 = d + 2.0f;
  float d2 = d + 1.0f;
  dinv1[i] = d1 > 0.0f ? rsqrtf(d1) : 0.0f;
  dinv[i]  = d2 > 0.0f ? rsqrtf(d2) : 0.0f;
}

// ---- per-(edge|self-loop) normalized weights -------------------------------
__global__ void k_norm(const int* __restrict__ src, const int* __restrict__ dst,
                       const float* __restrict__ ew,
                       const float* __restrict__ dinv1, const float* __restrict__ dinv,
                       float* __restrict__ norm1, float* __restrict__ norm_,
                       int E, int N) {
  int e = blockIdx.x * blockDim.x + threadIdx.x;
  if (e < E) {
    int s = src[e], d = dst[e];
    float w = ew[e];
    norm1[e] = dinv1[s] * w * dinv1[d];
    norm_[e] = dinv[s]  * w * dinv[d];
  } else if (e < E + N) {
    int i = e - E;
    float a = dinv1[i], b = dinv[i];
    norm1[e] = a * 2.0f * a;   // improved=True -> fill 2.0
    norm_[e] = b * 1.0f * b;
  }
}

// ---- prep: W (f32 [k][n]) -> bf16 transposed [n][k] ------------------------
__global__ void k_prep_w(const float* __restrict__ W, __bf16* __restrict__ Wt) {
  int i = blockIdx.x * blockDim.x + threadIdx.x;
  if (i >= 128 * 128) return;
  int k = i >> 7, n = i & 127;
  Wt[n * 128 + k] = (__bf16)W[i];
}

// ---- prep: f32 -> bf16 elementwise -----------------------------------------
__global__ void k_f2bf(const float* __restrict__ in, __bf16* __restrict__ out,
                       long long n) {
  long long i = (long long)blockIdx.x * blockDim.x + threadIdx.x;
  if (i < n) out[i] = (__bf16)in[i];
}

// ---- WMMA GEMM: T[rows,128] = Xbf[rows,128] @ Wt^T (bf16 in, f32 acc) ------
#define GEMM_ROWS 64
__global__ __launch_bounds__(256) void k_gemm128(const __bf16* __restrict__ Xbf,
                                                 const __bf16* __restrict__ Wt,
                                                 float* __restrict__ T, int rows) {
  extern __shared__ __bf16 smem[];          // 48 KB dynamic LDS
  __bf16* sWt = smem;                       // offset 0     : 128x128 [n][k]
  __bf16* sA  = smem + 128 * 128;           // offset 32768 : 64x128  [m][k]
  const int tid  = threadIdx.x;
  const int wave = tid >> 5;
  const int lane = tid & 31;
  const int row_base = blockIdx.x * GEMM_ROWS;

  // TDM: wave 0 issues both tile loads (in-order per wave), waits, barrier.
  if (wave == 0) {
    tdm_load_2d_bf16(0u, Wt, 128u, 128u);
    tdm_load_2d_bf16(32768u, Xbf + (size_t)row_base * 128,
                     (unsigned)(rows - row_base), (unsigned)GEMM_ROWS);
    __builtin_amdgcn_s_wait_tensorcnt(0);
  }
  __syncthreads();

  // A 16x32 bf16 (ISA 7.12.2): M = lane&15; lanes<16 K 0..7/16..23,
  // lanes>=16 K 8..15/24..31  -> two contiguous 8-elt runs.
  // B 32x16 bf16: N = lane&15; lanes<16 K 0..7/16..23, lanes>=16 K 8..15/24..31
  //   -> two contiguous 8-elt runs in sWt.
  const int m     = lane & 15;
  const int akb   = (lane >> 4) << 3;       // 0 or 8
  const int bn    = lane & 15;
  const int bkh   = (lane >> 4) << 4;       // 0 or 16
  const int mbase = (lane >> 4) << 3;       // D: M = mbase + r
  const int col_base = wave * 16;

  // load all 4 B chunk-fragments once (kept in registers, reused 4x)
  v16bf bfrag[4];
#pragma unroll
  for (int kc = 0; kc < 4; ++kc)
    bfrag[kc] = cat8(*(const v8bf*)&sWt[bn * 128 + kc * 32 + akb],
                     *(const v8bf*)&sWt[bn * 128 + kc * 32 + 16 + akb]);

#pragma unroll
  for (int rt = 0; rt < GEMM_ROWS / 16; ++rt) {
    const int tile_row = row_base + rt * 16;
    if (tile_row >= rows) break;            // uniform: tile fully out of range

    const int ml = rt * 16 + m;
    v8f acc = {};
#pragma unroll
    for (int kc = 0; kc < 4; ++kc) {
      const int k0 = kc * 32;
      v8bf alo = *(const v8bf*)&sA[ml * 128 + k0 + akb];
      v8bf ahi = *(const v8bf*)&sA[ml * 128 + k0 + 16 + akb];
      acc = __builtin_amdgcn_wmma_f32_16x16x32_bf16(false, cat8(alo, ahi),
                                                    false, bfrag[kc],
                                                    (short)0, acc, false, false);
    }

    float* p = T + (size_t)(tile_row + mbase) * 128 + col_base + bn;
    if (tile_row + 16 <= rows) {
      // uniform fast path: 8 unguarded stores at immediate offsets
#pragma unroll
      for (int r = 0; r < 8; ++r) p[(size_t)r * 128] = acc[r];
    } else {
#pragma unroll
      for (int r = 0; r < 8; ++r)
        if (tile_row + mbase + r < rows) p[(size_t)r * 128] = acc[r];
    }
  }
}

// ---- edge scatter: ACC[dst] += T[src] * nrm, wave per edge, 4 floats/lane --
__global__ __launch_bounds__(256) void k_scatter(const float* __restrict__ T,
                                                 const int* __restrict__ src,
                                                 const int* __restrict__ dst,
                                                 const float* __restrict__ nrm,
                                                 float* __restrict__ ACC,
                                                 int E, int N) {
  const long long gid  = (long long)blockIdx.x * blockDim.x + threadIdx.x;
  const long long edge = gid >> 5;
  const int lane       = (int)(gid & 31);
  const long long total = (long long)E + N;
  if (edge >= total) return;
  int s, d;
  float w = nrm[edge];
  if (edge < E) { s = src[edge]; d = dst[edge]; }
  else          { s = d = (int)(edge - E); }
  const float4 v = *((const float4*)(T + (size_t)s * 128) + lane);
  float* ap = ACC + (size_t)d * 128 + lane * 4;
  fatomic_add(ap + 0, v.x * w);
  fatomic_add(ap + 1, v.y * w);
  fatomic_add(ap + 2, v.z * w);
  fatomic_add(ap + 3, v.w * w);
}

// ---- bias + optional ReLU; writes f32 and bf16 copies ----------------------
__global__ void k_bias_act(const float* __restrict__ ACC, const float* __restrict__ b,
                           float* __restrict__ H, __bf16* __restrict__ Hbf,
                           long long n, int relu) {
  long long i = (long long)blockIdx.x * blockDim.x + threadIdx.x;
  if (i >= n) return;
  float v = ACC[i] + b[i & 127];
  v = (relu && v < 0.0f) ? 0.0f : v;
  H[i]   = v;
  Hbf[i] = (__bf16)v;
}

// ---- mean-pool accumulation -------------------------------------------------
__global__ void k_pool(const float* __restrict__ H, const int* __restrict__ batch,
                       float* __restrict__ pooled, float* __restrict__ cnt, int N) {
  long long i = (long long)blockIdx.x * blockDim.x + threadIdx.x;
  if (i >= (long long)N * 128) return;
  int node = (int)(i >> 7);
  int f    = (int)(i & 127);
  int g    = batch[node];
  fatomic_add(&pooled[(size_t)g * 128 + f], H[i]);
  if (f == 0) fatomic_add(&cnt[g], 1.0f);
}

// ---- head: out[b,c] = (pooled_sum[b,:]/max(cnt,1)) . Wlin[:,c] + blin[c] ---
__global__ void k_final(const float* __restrict__ pooled, const float* __restrict__ cnt,
                        const float* __restrict__ Wlin, const float* __restrict__ blin,
                        float* __restrict__ out, int B, int C) {
  int i = blockIdx.x * blockDim.x + threadIdx.x;
  if (i >= B * C) return;
  int b = i / C, c = i % C;
  float s = 0.0f;
#pragma unroll 4
  for (int k = 0; k < 128; ++k) s += pooled[(size_t)b * 128 + k] * Wlin[k * C + c];
  float n = cnt[b];
  out[i] = s / (n > 1.0f ? n : 1.0f) + blin[c];
}

// ---------------------------------------------------------------------------
extern "C" void kernel_launch(void* const* d_in, const int* in_sizes, int n_in,
                              void* d_out, int out_size, void* d_ws, size_t ws_size,
                              hipStream_t stream) {
  const float* x    = (const float*)d_in[0];
  const int*   ei   = (const int*)  d_in[1];
  const int*   batch= (const int*)  d_in[2];
  const float* ew   = (const float*)d_in[3];
  const float* W1   = (const float*)d_in[4];
  const float* b1   = (const float*)d_in[5];
  const float* W2   = (const float*)d_in[6];
  const float* b2   = (const float*)d_in[7];
  const float* W3   = (const float*)d_in[8];
  const float* b3   = (const float*)d_in[9];
  const float* Wlin = (const float*)d_in[10];
  const float* blin = (const float*)d_in[11];
  float* out = (float*)d_out;

  const int N = in_sizes[2];           // 50000
  const int E = in_sizes[3];           // 1600000
  const int C = in_sizes[11];          // 10
  const int B = out_size / C;          // 512
  const int H = 128;

  const int* src = ei;
  const int* dst = ei + E;

  // workspace carve-up
  float* ws    = (float*)d_ws;
  float* deg   = ws;                         // N
  float* dinv1 = deg + N;                    // N
  float* dinv  = dinv1 + N;                  // N
  float* norm1 = dinv + N;                   // E+N
  float* norm_ = norm1 + (size_t)(E + N);    // E+N
  float* T     = norm_ + (size_t)(E + N);    // N*H
  float* ACC   = T + (size_t)N * H;          // N*H
  float* Hb    = ACC + (size_t)N * H;        // N*H
  float* pooled= Hb + (size_t)N * H;         // B*H
  float* cnt   = pooled + (size_t)B * H;     // B
  __bf16* BF   = (__bf16*)(cnt + B);         // N*H bf16 (GEMM input)
  __bf16* Wt   = BF + (size_t)N * H;         // 3 * 128*128 bf16 (transposed)

  const int TPB = 256;

  // --- normalization prep ---
  (void)hipMemsetAsync(deg, 0, (size_t)N * sizeof(float), stream);
  k_deg <<<cdiv(E, TPB), TPB, 0, stream>>>(dst, ew, deg, E);
  k_dinv<<<cdiv(N, TPB), TPB, 0, stream>>>(deg, dinv1, dinv, N);
  k_norm<<<cdiv(E + N, TPB), TPB, 0, stream>>>(src, dst, ew, dinv1, dinv,
                                               norm1, norm_, E, N);

  // --- bf16 operand prep ---
  const long long NH = (long long)N * H;
  const int ew_grid  = (int)((NH + TPB - 1) / TPB);
  k_prep_w<<<cdiv(128 * 128, TPB), TPB, 0, stream>>>(W1, Wt);
  k_prep_w<<<cdiv(128 * 128, TPB), TPB, 0, stream>>>(W2, Wt + 16384);
  k_prep_w<<<cdiv(128 * 128, TPB), TPB, 0, stream>>>(W3, Wt + 32768);
  k_f2bf  <<<ew_grid, TPB, 0, stream>>>(x, BF, NH);

  const int gemm_grid    = cdiv(N, GEMM_ROWS);
  const long long sc_thr = ((long long)(E + N)) * 32;
  const int sc_grid      = (int)((sc_thr + TPB - 1) / TPB);
  const size_t gemm_lds  = (size_t)(128 * 128 + GEMM_ROWS * 128) * sizeof(__bf16);

  const float* bs[3] = { b1, b2, b3 };
  for (int layer = 0; layer < 3; ++layer) {
    const float* nrm = (layer == 0) ? norm1 : norm_;
    k_gemm128<<<gemm_grid, TPB, gemm_lds, stream>>>(BF, Wt + layer * 16384, T, N);
    (void)hipMemsetAsync(ACC, 0, (size_t)NH * sizeof(float), stream);
    k_scatter<<<sc_grid, TPB, 0, stream>>>(T, src, dst, nrm, ACC, E, N);
    k_bias_act<<<ew_grid, TPB, 0, stream>>>(ACC, bs[layer], Hb, BF, NH,
                                            layer < 2 ? 1 : 0);
  }

  // --- pool + head ---
  (void)hipMemsetAsync(pooled, 0, ((size_t)B * H + B) * sizeof(float), stream);
  k_pool <<<ew_grid, TPB, 0, stream>>>(Hb, batch, pooled, cnt, N);
  k_final<<<cdiv(B * C, TPB), TPB, 0, stream>>>(pooled, cnt, Wlin, blin, out, B, C);
}
